// DMPLRPPoolLayer_68049461838036
// MI455X (gfx1250) — matrix operand order repaired and checked by reading, not compiled, not run on hardware
//
#include <hip/hip_runtime.h>
#include <hip/hip_bf16.h>

// ---------------------------------------------------------------------------
// Problem constants (from the reference)
// ---------------------------------------------------------------------------
#define NN     20000
#define EE     640000
#define DIMC   128
#define LLQ    16          // L*L
#define NPERMC 20000
#define BBC    256
#define RRC    (NPERMC * LLQ)
#define EPSC   1e-5f
#define LSTRIDE 132        // LDS row stride in floats (16B aligned rows)

typedef __bf16 bf16_t;
typedef __attribute__((ext_vector_type(16))) __bf16 v16bf;
typedef __attribute__((ext_vector_type(8)))  float  v8f;

#define V8F_ZERO {0.f,0.f,0.f,0.f,0.f,0.f,0.f,0.f}

// weight fragment slots (each 4*8*32*16 = 16384 bf16 elems = 32 KB)
#define WF_DST   0
#define WF_SRC   1
#define WF_OUT   2
#define WF_IN    3
#define WF_SD    4
#define WF_ELOOP 5
#define WF_NLOOP 6
#define WF_NW1   7
#define WF_NW2   8
#define WF_EW1   9
#define WF_EW2   10
#define WF_ELEMS 16384
#define LRP_ELEMS (64*8*32*16)   // 64 K-chunks for the 2048x128 flat lrp weight

// ---------------------------------------------------------------------------
// Workspace layout (bytes).  Total requirement ~191 MB.
// ---------------------------------------------------------------------------
static constexpr size_t OFF_WFRAG = 0;                       // 11*16384*2 = 360448
static constexpr size_t OFF_LRPF  = 360448;                  // 262144*2   = 524288
static constexpr size_t OFF_DEG   = 884736;                  // NN*4       = 80000
static constexpr size_t OFF_AGG   = 965120;                  // NN*128*4   = 10240000
static constexpr size_t OFF_NH    = 11205120;                // NN*128*2   = 5120000
static constexpr size_t OFF_NOUT  = 16325120;                // NN*128*4   = 10240000
static constexpr size_t OFF_STATS = 26565120;                // 1024 floats
static constexpr size_t OFF_HE    = 26569216;                // EE*128*2   = 163840000
// stats float layout: [0]nsum [128]nsumsq [256]esum [384]esumsq
//                     [512]nscale [640]nshift [768]escale [896]eshift

// ---------------------------------------------------------------------------
// WMMA helpers
// ---------------------------------------------------------------------------
__device__ __forceinline__ v8f wmma_bf16(v16bf a, v16bf b, v8f c) {
  return __builtin_amdgcn_wmma_f32_16x16x32_bf16(false, a, false, b, (short)0, c,
                                                 false, false);
}

// A fragment (16x32 bf16) from an LDS row (row-major fp32 tile).
// ISA layout: lanes 0-15 rows M=0..15 with K groups {0-7,16-23},
//             lanes 16-31 same rows with K groups {8-15,24-31}.
__device__ __forceinline__ v16bf afrag_from_lds(const float* row, int kchunk, int hi) {
  v16bf f;
  const int base = kchunk * 32 + hi * 8;
#pragma unroll
  for (int v = 0; v < 8; ++v) {
    int K = (v < 4) ? (base + 2 * v) : (base + 16 + 2 * (v - 4));
    f[2 * v]     = (__bf16)row[K];
    f[2 * v + 1] = (__bf16)row[K + 1];
  }
  return f;
}

// B fragment: pre-swizzled in ws -> one aligned 32-byte per-lane load.
__device__ __forceinline__ v16bf load_bfrag(const bf16_t* frag, int k, int j, int lane) {
  return *(const v16bf*)(frag + ((size_t)((k * 8 + j) * 32 + lane) << 4));
}

// Cooperative staging (VGPR path): 32 lanes stage 16 rows x 128 f32 into LDS.
// lane -> row (lane&15), column half (lane>>4)*64.
__device__ __forceinline__ void stage_row16(float* stage, const float* row, int lane) {
  const int M = lane & 15, hi = lane >> 4;
  const float4* s = (const float4*)row + hi * 16;
  float* d = stage + M * LSTRIDE + hi * 64;
#pragma unroll
  for (int i = 0; i < 16; ++i) {
    float4 t = s[i];
    d[4 * i + 0] = t.x; d[4 * i + 1] = t.y; d[4 * i + 2] = t.z; d[4 * i + 3] = t.w;
  }
}

// Cooperative staging (CDNA5 async path): each lane issues 16 b128 async
// copies global->LDS, bypassing VGPRs.  INST_OFFSET is added to both the LDS
// and global addresses, so one base pair + immediate offsets covers the
// contiguous 256B slice.  Generic LDS pointers truncate to the wave-relative
// LDS byte offset in the low 32 bits (ISA 10.2 aperture mapping).
__device__ __forceinline__ void async_stage_row16(float* stage, const float* row,
                                                  int lane) {
  const int M = lane & 15, hi = lane >> 4;
  const void* g = (const void*)(row + hi * 64);
  unsigned l = (unsigned)(unsigned long long)(const void*)(stage + M * LSTRIDE + hi * 64);
#pragma unroll
  for (int i = 0; i < 16; ++i)
    asm volatile("global_load_async_to_lds_b128 %0, %1, off offset:%2"
                 :: "v"(l), "v"(g), "i"(i * 16) : "memory");
}

__device__ __forceinline__ void wait_async0() {
  asm volatile("s_wait_asynccnt 0x0" ::: "memory");
}

// ---------------------------------------------------------------------------
// K0: zero scratch + pooled output region
// ---------------------------------------------------------------------------
__global__ void k0_zero(float* agg, int* deg, float* stats, float* pooled) {
  int i = blockIdx.x * blockDim.x + threadIdx.x;
  if (i < NN * DIMC) agg[i] = 0.f;
  if (i < NN)        deg[i] = 0;
  if (i < 1024)      stats[i] = 0.f;
  if (i < BBC * DIMC) pooled[i] = 0.f;
}

// ---------------------------------------------------------------------------
// Kw: convert weights to bf16 B-fragments (incl. src_w - dst_w combo and
//     the flattened (a*128+b, c) lrp weight).
// ---------------------------------------------------------------------------
__global__ void kw_prep(const float* __restrict__ in_w,  const float* __restrict__ out_w,
                        const float* __restrict__ src_w, const float* __restrict__ dst_w,
                        const float* __restrict__ nloop_w, const float* __restrict__ eloop_w,
                        const float* __restrict__ lrp_w,
                        const float* __restrict__ nw1, const float* __restrict__ nw2,
                        const float* __restrict__ ew1, const float* __restrict__ ew2,
                        bf16_t* __restrict__ wfrag, bf16_t* __restrict__ lrpfrag) {
  int t = blockIdx.x * blockDim.x + threadIdx.x;
  if (t < 11 * WF_ELEMS) {
    int w = t / WF_ELEMS, r = t % WF_ELEMS;
    int e = r & 15, lane = (r >> 4) & 31, j = (r >> 9) & 7, k = r >> 12;
    int K = k * 32 + ((lane >= 16) ? 8 : 0) + e + ((e >= 8) ? 8 : 0);
    int n = j * 16 + (lane & 15);
    int idx = K * DIMC + n;
    float val;
    switch (w) {
      case WF_DST:   val = dst_w[idx]; break;
      case WF_SRC:   val = src_w[idx]; break;
      case WF_OUT:   val = out_w[idx]; break;
      case WF_IN:    val = in_w[idx]; break;
      case WF_SD:    val = src_w[idx] - dst_w[idx]; break;
      case WF_ELOOP: val = eloop_w[idx]; break;
      case WF_NLOOP: val = nloop_w[idx]; break;
      case WF_NW1:   val = nw1[idx]; break;
      case WF_NW2:   val = nw2[idx]; break;
      case WF_EW1:   val = ew1[idx]; break;
      default:       val = ew2[idx]; break;
    }
    wfrag[t] = (bf16_t)val;
  }
  int t2 = t - 11 * WF_ELEMS;
  if (t2 >= 0 && t2 < LRP_ELEMS) {
    int e = t2 & 15, lane = (t2 >> 4) & 31, j = (t2 >> 9) & 7, ka = t2 >> 12;
    int K = ka * 32 + ((lane >= 16) ? 8 : 0) + e + ((e >= 8) ? 8 : 0);   // K in [0,2048): K = a*128 + b
    int a = K >> 7, b = K & 127;
    int c = j * 16 + (lane & 15);
    lrpfrag[t2] = (bf16_t)lrp_w[((size_t)b * DIMC + c) * LLQ + a];
  }
}

// ---------------------------------------------------------------------------
// K1: out-degree histogram of src
// ---------------------------------------------------------------------------
__global__ void k1_deg(const int* __restrict__ src, int* __restrict__ deg) {
  int i = blockIdx.x * blockDim.x + threadIdx.x;
  if (i < EE) atomicAdd(&deg[src[i]], 1);
}

// ---------------------------------------------------------------------------
// K2: fused edge phase.  Per 16-edge tile (1 wave):
//   stage a/b/edge rows via async global->LDS b128 copies (ASYNCcnt path)
//   edge_msg = a@dst_w - b@src_w            (a,b gathered per rev flag)
//   node_msg = rev ? ef@out_w : -(ef@in_w)  -> atomic scatter into node_agg
//   e_pre = ef@eloop_w + 2(1+dlog)*ef@(src_w-dst_w) + edge_msg + ebias
//   h_e   = e_pre @ emlp_w1 + b1  (stored bf16) + channel sum/sumsq atomics
// ---------------------------------------------------------------------------
__global__ __launch_bounds__(128) void k2_edge(
    const float* __restrict__ node_feat, const float* __restrict__ edge_feat,
    const int* __restrict__ src, const int* __restrict__ dst,
    const unsigned char* __restrict__ rev, const int* __restrict__ deg,
    const float* __restrict__ ebias, const float* __restrict__ emlp_b1,
    const bf16_t* __restrict__ wfrag,
    float* __restrict__ node_agg, bf16_t* __restrict__ h_e,
    float* __restrict__ stats) {
  __shared__ float smem[4][3][16 * LSTRIDE];
  const int lane = threadIdx.x & 31;
  const int wave = threadIdx.x >> 5;
  float* stA = smem[wave][0];
  float* stB = smem[wave][1];
  float* stE = smem[wave][2];
  const int tile = blockIdx.x * 4 + wave;
  const int e0 = tile * 16;
  const int M = lane & 15, hi = lane >> 4;

  // gather meta for the row this lane stages
  const int eM = e0 + M;
  const int sM = src[eM], dM = dst[eM];
  const bool rvM = rev[eM] != 0;

  // burst all three tiles through the async LDS path (48 ops <= ASYNCcnt max)
  async_stage_row16(stA, node_feat + (size_t)(rvM ? sM : dM) * DIMC, lane);
  async_stage_row16(stB, node_feat + (size_t)(rvM ? dM : sM) * DIMC, lane);
  async_stage_row16(stE, edge_feat + (size_t)eM * DIMC, lane);

  // per-lane meta for its 8 C/D rows (M = v + 8*hi) -- overlaps async copies
  bool  revv[8]; int dstv[8]; float dlogv[8];
#pragma unroll
  for (int v = 0; v < 8; ++v) {
    int r = e0 + hi * 8 + v;
    revv[v] = rev[r] != 0;
    int dd = dst[r];
    dstv[v] = dd;
    dlogv[v] = log2f(1.0f + (float)deg[dd]);
  }

  wait_async0();

  v16bf aA[4], bA[4], eA[4];
#pragma unroll
  for (int k = 0; k < 4; ++k) aA[k] = afrag_from_lds(stA + M * LSTRIDE, k, hi);
#pragma unroll
  for (int k = 0; k < 4; ++k) bA[k] = afrag_from_lds(stB + M * LSTRIDE, k, hi);
#pragma unroll
  for (int k = 0; k < 4; ++k) eA[k] = afrag_from_lds(stE + M * LSTRIDE, k, hi);

  const bf16_t* WDST = wfrag + (size_t)WF_DST   * WF_ELEMS;
  const bf16_t* WSRC = wfrag + (size_t)WF_SRC   * WF_ELEMS;
  const bf16_t* WOUT = wfrag + (size_t)WF_OUT   * WF_ELEMS;
  const bf16_t* WIN  = wfrag + (size_t)WF_IN    * WF_ELEMS;
  const bf16_t* WSD  = wfrag + (size_t)WF_SD    * WF_ELEMS;
  const bf16_t* WEL  = wfrag + (size_t)WF_ELOOP * WF_ELEMS;

  for (int j = 0; j < 8; ++j) {
    v8f em = V8F_ZERO, es = V8F_ZERO, mo = V8F_ZERO,
        mi = V8F_ZERO, gg = V8F_ZERO, el = V8F_ZERO;
#pragma unroll
    for (int k = 0; k < 4; ++k) {
      em = wmma_bf16(aA[k], load_bfrag(WDST, k, j, lane), em);
      es = wmma_bf16(bA[k], load_bfrag(WSRC, k, j, lane), es);
      mo = wmma_bf16(eA[k], load_bfrag(WOUT, k, j, lane), mo);
      mi = wmma_bf16(eA[k], load_bfrag(WIN,  k, j, lane), mi);
      gg = wmma_bf16(eA[k], load_bfrag(WSD,  k, j, lane), gg);
      el = wmma_bf16(eA[k], load_bfrag(WEL,  k, j, lane), el);
    }
    const int cN = j * 16 + M;
    const float eb = ebias[cN];
#pragma unroll
    for (int v = 0; v < 8; ++v) {
      float nm = revv[v] ? mo[v] : -mi[v];
      atomicAdd(&node_agg[(size_t)dstv[v] * DIMC + cN], nm);
      float ep = el[v] + 2.0f * (1.0f + dlogv[v]) * gg[v] + (em[v] - es[v]) + eb;
      stA[(hi * 8 + v) * LSTRIDE + cN] = ep;   // reuse tile A for e_pre
    }
  }

  // h_e = e_pre @ emlp_w1 + b1
  v16bf pA[4];
#pragma unroll
  for (int k = 0; k < 4; ++k) pA[k] = afrag_from_lds(stA + M * LSTRIDE, k, hi);
  const bf16_t* WW1 = wfrag + (size_t)WF_EW1 * WF_ELEMS;
  for (int j = 0; j < 8; ++j) {
    v8f hh = V8F_ZERO;
#pragma unroll
    for (int k = 0; k < 4; ++k) hh = wmma_bf16(pA[k], load_bfrag(WW1, k, j, lane), hh);
    const int cN = j * 16 + M;
    const float b1 = emlp_b1[cN];
    float ps = 0.f, ps2 = 0.f;
#pragma unroll
    for (int v = 0; v < 8; ++v) {
      float h = hh[v] + b1;
      h_e[(size_t)(e0 + hi * 8 + v) * DIMC + cN] = (bf16_t)h;
      ps += h; ps2 += h * h;
    }
    atomicAdd(&stats[256 + cN], ps);
    atomicAdd(&stats[384 + cN], ps2);
  }
}

// ---------------------------------------------------------------------------
// K3a: node phase.  x = node_feat@nloop_w + node_agg + nbias; h_n = x@nmlp_w1+b1
// ---------------------------------------------------------------------------
__global__ __launch_bounds__(32) void k3a_node(
    const float* __restrict__ node_feat, const float* __restrict__ node_agg,
    const float* __restrict__ nbias, const float* __restrict__ nmlp_b1,
    const bf16_t* __restrict__ wfrag, bf16_t* __restrict__ n_h,
    float* __restrict__ stats) {
  __shared__ float stage[16 * LSTRIDE];
  const int lane = threadIdx.x;
  const int M = lane & 15, hi = lane >> 4;
  const int r0 = blockIdx.x * 16;

  async_stage_row16(stage, node_feat + (size_t)(r0 + M) * DIMC, lane);
  wait_async0();
  v16bf fA[4];
#pragma unroll
  for (int k = 0; k < 4; ++k) fA[k] = afrag_from_lds(stage + M * LSTRIDE, k, hi);

  const bf16_t* WNL = wfrag + (size_t)WF_NLOOP * WF_ELEMS;
  for (int j = 0; j < 8; ++j) {
    v8f acc = V8F_ZERO;
#pragma unroll
    for (int k = 0; k < 4; ++k) acc = wmma_bf16(fA[k], load_bfrag(WNL, k, j, lane), acc);
    const int cN = j * 16 + M;
    const float nb = nbias[cN];
#pragma unroll
    for (int v = 0; v < 8; ++v) {
      int r = r0 + hi * 8 + v;
      stage[(hi * 8 + v) * LSTRIDE + cN] = acc[v] + node_agg[(size_t)r * DIMC + cN] + nb;
    }
  }

  v16bf xA[4];
#pragma unroll
  for (int k = 0; k < 4; ++k) xA[k] = afrag_from_lds(stage + M * LSTRIDE, k, hi);
  const bf16_t* WN1 = wfrag + (size_t)WF_NW1 * WF_ELEMS;
  for (int j = 0; j < 8; ++j) {
    v8f hh = V8F_ZERO;
#pragma unroll
    for (int k = 0; k < 4; ++k) hh = wmma_bf16(xA[k], load_bfrag(WN1, k, j, lane), hh);
    const int cN = j * 16 + M;
    const float b1 = nmlp_b1[cN];
    float ps = 0.f, ps2 = 0.f;
#pragma unroll
    for (int v = 0; v < 8; ++v) {
      float h = hh[v] + b1;
      n_h[(size_t)(r0 + hi * 8 + v) * DIMC + cN] = (bf16_t)h;
      ps += h; ps2 += h * h;
    }
    atomicAdd(&stats[0 + cN], ps);
    atomicAdd(&stats[128 + cN], ps2);
  }
}

// ---------------------------------------------------------------------------
// Kstats: finalize batchnorm scale/shift for nodes and edges
// ---------------------------------------------------------------------------
__global__ void kstats(float* __restrict__ stats,
                       const float* __restrict__ ng, const float* __restrict__ nbeta,
                       const float* __restrict__ eg, const float* __restrict__ ebeta) {
  int t = threadIdx.x;
  if (t < 128) {
    float mu  = stats[t] / (float)NN;
    float var = stats[128 + t] / (float)NN - mu * mu;
    float s = ng[t] * rsqrtf(var + EPSC);
    stats[512 + t] = s;
    stats[640 + t] = nbeta[t] - mu * s;
  } else if (t < 256) {
    int c = t - 128;
    float mu  = stats[256 + c] / (float)EE;
    float var = stats[384 + c] / (float)EE - mu * mu;
    float s = eg[c] * rsqrtf(var + EPSC);
    stats[768 + c] = s;
    stats[896 + c] = ebeta[c] - mu * s;
  }
}

// ---------------------------------------------------------------------------
// K_mlp2: out = relu(h*scale + shift) @ w2 + b2   (nodes and edges share this)
// ---------------------------------------------------------------------------
__global__ void k_mlp2(const bf16_t* __restrict__ h, const float* __restrict__ scale,
                       const float* __restrict__ shift, const float* __restrict__ b2,
                       const bf16_t* __restrict__ w2frag, float* __restrict__ out) {
  extern __shared__ float smem[];
  const int lane = threadIdx.x & 31;
  const int wave = threadIdx.x >> 5;
  float* stage = smem + (size_t)wave * (16 * LSTRIDE);
  const int tile = blockIdx.x * (blockDim.x >> 5) + wave;
  const int r0 = tile * 16;
  const int M = lane & 15, hi = lane >> 4;

  const bf16_t* hrow = h + (size_t)(r0 + M) * DIMC + hi * 64;
  float* srow = stage + M * LSTRIDE + hi * 64;
#pragma unroll
  for (int i = 0; i < 64; ++i) {
    int c = hi * 64 + i;
    float v = (float)hrow[i];
    v = v * scale[c] + shift[c];
    srow[i] = v > 0.f ? v : 0.f;
  }

  v16bf A[4];
#pragma unroll
  for (int k = 0; k < 4; ++k) A[k] = afrag_from_lds(stage + M * LSTRIDE, k, hi);
  for (int j = 0; j < 8; ++j) {
    v8f acc = V8F_ZERO;
#pragma unroll
    for (int k = 0; k < 4; ++k) acc = wmma_bf16(A[k], load_bfrag(w2frag, k, j, lane), acc);
    const int cN = j * 16 + M;
    const float bb = b2[cN];
#pragma unroll
    for (int v = 0; v < 8; ++v)
      out[(size_t)(r0 + hi * 8 + v) * DIMC + cN] = acc[v] + bb;
  }
}

// ---------------------------------------------------------------------------
// K5: perm build + flat LRP contraction + pooled scatter.
//   lrp[d,c] = sum_{a,b} perm[d,a,b] * lrp_w[b,c,a]  (flat 2048-K matmul)
//   pooled[pool_rows[d], c] += pool_vals[d] * (lrp[d,c] + lrp_bias[c])
// ---------------------------------------------------------------------------
__global__ __launch_bounds__(32) void k5_lrp(
    const float* __restrict__ nout, const float* __restrict__ eout,
    const int* __restrict__ n2p_cols, const float* __restrict__ n2p_vals,
    const int* __restrict__ e2p_cols, const float* __restrict__ e2p_vals,
    const int* __restrict__ pool_rows, const float* __restrict__ pool_vals,
    const float* __restrict__ lrp_bias, const bf16_t* __restrict__ lrpfrag,
    float* __restrict__ pooled) {
  __shared__ float stage[16 * LSTRIDE];
  const int lane = threadIdx.x;
  const int M = lane & 15, hi = lane >> 4;
  const int d0 = blockIdx.x * 16;

  v8f acc[8];
#pragma unroll
  for (int j = 0; j < 8; ++j) acc[j] = V8F_ZERO;

  for (int a = 0; a < LLQ; ++a) {
    // stage perm rows for this 'a': row d0+M, cols [hi*64, hi*64+64)
    const int r = (d0 + M) * LLQ + a;
    const int nc = n2p_cols[r], ec = e2p_cols[r];
    const float nv = n2p_vals[r], ev = e2p_vals[r];
    const float* np = nout + (size_t)nc * DIMC + hi * 64;
    const float* ep = eout + (size_t)ec * DIMC + hi * 64;
    float* drow = stage + M * LSTRIDE + hi * 64;
#pragma unroll
    for (int i = 0; i < 64; ++i) drow[i] = nv * np[i] + ev * ep[i];

    v16bf pA[4];
#pragma unroll
    for (int kc = 0; kc < 4; ++kc) pA[kc] = afrag_from_lds(stage + M * LSTRIDE, kc, hi);
#pragma unroll
    for (int kc = 0; kc < 4; ++kc) {
      const int ka = a * 4 + kc;
#pragma unroll
      for (int j = 0; j < 8; ++j)
        acc[j] = wmma_bf16(pA[kc], load_bfrag(lrpfrag, ka, j, lane), acc[j]);
    }
  }

  int prow[8]; float pv[8];
#pragma unroll
  for (int v = 0; v < 8; ++v) {
    int d = d0 + hi * 8 + v;
    prow[v] = pool_rows[d];
    pv[v] = pool_vals[d];
  }
#pragma unroll
  for (int j = 0; j < 8; ++j) {
    const int cN = j * 16 + M;
    const float lb = lrp_bias[cN];
#pragma unroll
    for (int v = 0; v < 8; ++v)
      atomicAdd(&pooled[(size_t)prow[v] * DIMC + cN], pv[v] * (acc[j][v] + lb));
  }
}

// ---------------------------------------------------------------------------
// Host launch
// ---------------------------------------------------------------------------
extern "C" void kernel_launch(void* const* d_in, const int* in_sizes, int n_in,
                              void* d_out, int out_size, void* d_ws, size_t ws_size,
                              hipStream_t stream) {
  (void)in_sizes; (void)n_in; (void)out_size; (void)ws_size;

  const float* node_feat = (const float*)d_in[0];
  const float* edge_feat = (const float*)d_in[1];
  const int*   src       = (const int*)d_in[2];
  const int*   dst       = (const int*)d_in[3];
  const unsigned char* rev = (const unsigned char*)d_in[4];
  const int*   n2p_cols  = (const int*)d_in[5];
  const float* n2p_vals  = (const float*)d_in[6];
  const int*   e2p_cols  = (const int*)d_in[7];
  const float* e2p_vals  = (const float*)d_in[8];
  const int*   pool_rows = (const int*)d_in[9];
  const float* pool_vals = (const float*)d_in[10];
  const float* in_w      = (const float*)d_in[11];
  const float* out_w     = (const float*)d_in[12];
  const float* src_w     = (const float*)d_in[13];
  const float* dst_w     = (const float*)d_in[14];
  const float* nloop_w   = (const float*)d_in[15];
  const float* eloop_w   = (const float*)d_in[16];
  const float* lrp_w     = (const float*)d_in[17];
  const float* nbias     = (const float*)d_in[18];
  const float* ebias     = (const float*)d_in[19];
  const float* lrp_bias  = (const float*)d_in[20];
  const float* nmlp_w1   = (const float*)d_in[21];
  const float* nmlp_b1   = (const float*)d_in[22];
  const float* nmlp_g    = (const float*)d_in[23];
  const float* nmlp_beta = (const float*)d_in[24];
  const float* nmlp_w2   = (const float*)d_in[25];
  const float* nmlp_b2   = (const float*)d_in[26];
  const float* emlp_w1   = (const float*)d_in[27];
  const float* emlp_b1   = (const float*)d_in[28];
  const float* emlp_g    = (const float*)d_in[29];
  const float* emlp_beta = (const float*)d_in[30];
  const float* emlp_w2   = (const float*)d_in[31];
  const float* emlp_b2   = (const float*)d_in[32];

  char* ws = (char*)d_ws;
  bf16_t* wfrag   = (bf16_t*)(ws + OFF_WFRAG);
  bf16_t* lrpfrag = (bf16_t*)(ws + OFF_LRPF);
  int*    deg     = (int*)   (ws + OFF_DEG);
  float*  agg     = (float*) (ws + OFF_AGG);
  bf16_t* n_h     = (bf16_t*)(ws + OFF_NH);
  float*  noutp   = (float*) (ws + OFF_NOUT);
  float*  stats   = (float*) (ws + OFF_STATS);
  bf16_t* h_e     = (bf16_t*)(ws + OFF_HE);

  float* pooled = (float*)d_out;                   // B*DIM
  float* eout   = (float*)d_out + BBC * DIMC;      // E*DIM

  // K0: zero agg/deg/stats/pooled
  k0_zero<<<(NN * DIMC + 255) / 256, 256, 0, stream>>>(agg, deg, stats, pooled);

  // Kw: weight fragments (11*16384 + 262144 = 442368 threads)
  kw_prep<<<(11 * WF_ELEMS + LRP_ELEMS) / 256, 256, 0, stream>>>(
      in_w, out_w, src_w, dst_w, nloop_w, eloop_w, lrp_w,
      nmlp_w1, nmlp_w2, emlp_w1, emlp_w2, wfrag, lrpfrag);

  // K1: out-degree histogram
  k1_deg<<<EE / 256, 256, 0, stream>>>(src, deg);

  // K2: fused edge phase (40000 tiles, 4 waves/block, async LDS staging)
  k2_edge<<<(EE / 16) / 4, 128, 0, stream>>>(
      node_feat, edge_feat, src, dst, rev, deg, ebias, emlp_b1,
      wfrag, agg, h_e, stats);

  // K3a: node phase (1250 tiles, 1 wave each)
  k3a_node<<<NN / 16, 32, 0, stream>>>(node_feat, agg, nbias, nmlp_b1,
                                       wfrag, n_h, stats);

  // Kstats: batchnorm finalize
  kstats<<<1, 256, 0, stream>>>(stats, nmlp_g, nmlp_beta, emlp_g, emlp_beta);

  // MLP second halves
  k_mlp2<<<NN / 16, 32, 16 * LSTRIDE * sizeof(float), stream>>>(
      n_h, stats + 512, stats + 640, nmlp_b2,
      wfrag + (size_t)WF_NW2 * WF_ELEMS, noutp);
  k_mlp2<<<(EE / 16) / 8, 256, 8 * 16 * LSTRIDE * sizeof(float), stream>>>(
      h_e, stats + 768, stats + 896, emlp_b2,
      wfrag + (size_t)WF_EW2 * WF_ELEMS, eout);

  // K5: LRP contraction + pooling
  k5_lrp<<<NPERMC / 16, 32, 0, stream>>>(
      noutp, eout, n2p_cols, n2p_vals, e2p_cols, e2p_vals,
      pool_rows, pool_vals, lrp_bias, lrpfrag, pooled);
}
